// LocalAttention_23261542875267
// MI455X (gfx1250) — compile-verified, matrix operand order checked by reading, not run on hardware
//
#include <hip/hip_runtime.h>

typedef __attribute__((ext_vector_type(16))) _Float16 v16h;
typedef __attribute__((ext_vector_type(8)))  _Float16 v8h;
typedef __attribute__((ext_vector_type(8)))  float    v8f;
typedef __attribute__((ext_vector_type(4)))  int      v4i;
typedef __attribute__((ext_vector_type(4)))  unsigned int u32x4;
typedef __attribute__((ext_vector_type(8)))  int      i32x8;

#define PATCH 7
#define PPAD  8               // padded tokens per sequence
#define HEADS 8
#define CCH   256
#define INNER 512
#define TTOT  196
#define VDIM  25
#define TBLK  28
#define NSEQ  11200
#define GSEQ  16              // sequences per block
#define MR    128             // 16 seq * 8 padded rows = 8 exact M-tiles
#define NGRP  (NSEQ/GSEQ)     // 700 blocks

// LDS strides (f16 elems). XS=264: +8 pad keeps 16B alignment and rotates banks.
#define XS 264
#define QS 72
#define YS 260                // f32 staging stride

// LDS region offsets (bytes); all 16B aligned
#define OFF_X  0                              // x tile     128*264*2 = 67584
#define OFF_QL 67584                          // Q rows     128*72*2  = 18432
#define OFF_OL 86016                          // O rows     128*72*2  = 18432
#define OFF_KB 104448                         // K B-frags  8*2*32*16*2 = 16384
#define OFF_VB 120832                         // V B-frags  8*4*32*16*2 = 32768
#define OFF_PA 153600                         // P A-frags  8*32*16*2   = 8192
#define OFF_SS 161792                         // S scores   8*256*4     = 8192
#define OFF_WB 169984                         // per-head weight stage  = 131072
#define SMEM_BYTES (OFF_WB + 131072)          // 301056 B < 320KB/WGP
#define ZERO_BYTES (32768 + 8192)             // vb + pa static-zero init

#define HCHUNK 16384          // f16 elems per (head, matrix) packed chunk
#define HBYTES 131072         // bytes of packed weights per head (4 chunks)

// -------------------------------------------------------------------------
// Kernel 1: pack Wq/Wk/Wv/Wo into f16 WMMA B-fragments, grouped per head:
//   ws[h][mat][chunk]  (chunk = 16384 f16 = 32KB) so one head = 128KB contiguous
// Fragment element e of lane L: j=e>>1, half=e&1, n=L&15,
//   k = (j<4?0:16) + (L>=16?8:0) + 2*(j&3) + half   (mirror of 16-bit A layout)
// -------------------------------------------------------------------------
__global__ __launch_bounds__(256) void pack_weights_kernel(
    const float* __restrict__ Wq, const float* __restrict__ Wkv,
    const float* __restrict__ Wo, _Float16* __restrict__ pk)
{
  int idx = blockIdx.x * 256 + threadIdx.x;       // [0, 4*131072)
  int mat = idx >> 17;                            // 0=Q 1=K 2=V 3=Wo
  int rem = idx & 131071;
  int e    = rem & 15;
  int lane = (rem >> 4) & 31;
  int fc   = rem >> 9;                            // nt*KC + kc
  int KC   = (mat == 3) ? 16 : 8;
  int kc   = fc % KC;
  int nt   = fc / KC;
  int j = e >> 1, half = e & 1;
  int k = ((j < 4) ? 0 : 16) + ((lane >= 16) ? 8 : 0) + 2 * (j & 3) + half;
  int n = lane & 15;
  int ng = nt * 16 + n;
  int kg = kc * 32 + k;
  float val;
  if      (mat == 0) val = Wq[ng * CCH + kg];
  else if (mat == 1) val = Wkv[ng * CCH + kg];
  else if (mat == 2) val = Wkv[(INNER + ng) * CCH + kg];
  else               val = Wo[ng * INNER + kg];
  // destination: per-head grouped layout
  int dst;
  if (mat < 3) {                 // head = nt>>2, in-head col tile = nt&3
    dst = ((nt >> 2) * 4 + mat) * HCHUNK + (((nt & 3) * 8 + kc) << 9) + (lane << 4) + e;
  } else {                       // head = kc>>1, in-head k chunk = kc&1
    dst = ((kc >> 1) * 4 + 3) * HCHUNK + ((nt * 2 + (kc & 1)) << 9) + (lane << 4) + e;
  }
  pk[dst] = (_Float16)val;
}

__device__ __forceinline__ v8f wmma32(v16h a, v16h b, v8f c) {
  return __builtin_amdgcn_wmma_f32_16x16x32_f16(false, a, false, b,
                                                (short)0, c, false, false);
}

// -------------------------------------------------------------------------
// Kernel 2: fully fused local attention; all GEMM stages on WMMA, per-head
// weights staged global->LDS via the Tensor Data Mover (TDM).
// -------------------------------------------------------------------------
__global__ __launch_bounds__(256) void local_attn_kernel(
    const float* __restrict__ fmap, const _Float16* __restrict__ pack,
    const float* __restrict__ bo, float* __restrict__ out)
{
  extern __shared__ char smem[];
  _Float16* xl = (_Float16*)(smem + OFF_X);
  _Float16* ql = (_Float16*)(smem + OFF_QL);
  _Float16* ol = (_Float16*)(smem + OFF_OL);
  _Float16* kb = (_Float16*)(smem + OFF_KB);
  _Float16* vb = (_Float16*)(smem + OFF_VB);
  _Float16* pa = (_Float16*)(smem + OFF_PA);
  float*    sS = (float*)(smem + OFF_SS);
  const _Float16* ws = (const _Float16*)(smem + OFF_WB);
  float*    ystage = (float*)smem;

  const int tid  = threadIdx.x;
  const int wave = tid >> 5;
  const int lane = tid & 31;
  const int g    = blockIdx.x;
  const int n16  = lane & 15;
  const int mofs = (lane >= 16) ? 8 : 0;

  // sequence owned by this thread in the (de)interleave phases
  const int sl = tid & 15;
  const int s  = g * GSEQ + sl;
  const int vv = s % VDIM;
  const int tt = (s / VDIM) % TBLK;
  const int bb = s / (VDIM * TBLK);
  const long gbase = ((long)bb * CCH * TTOT + (long)tt * PATCH) * VDIM + vv;

  // zero the static-zero slots of vb (k>=16) and pa (masked/cross positions)
  for (int i = tid; i < ZERO_BYTES / 16; i += 256) {
    v4i z = {};
    *(v4i*)(smem + OFF_VB + i * 16) = z;
  }

  // stage x: fmap[b,ch,t*7+p,v] -> xl[sl*8+p][ch]; pad row p=7 zeroed
  for (int pass = 0; pass < 128; ++pass) {
    int pair = pass * 16 + (tid >> 4);       // pair = ch*8 + p
    int ch = pair >> 3, p = pair & 7;
    float x = 0.f;
    if (p < PATCH) x = fmap[gbase + (long)ch * (TTOT * VDIM) + p * VDIM];
    xl[(sl * PPAD + p) * XS + ch] = (_Float16)x;
  }

  v8f zero = {};
  v8f acc[16];
  #pragma unroll
  for (int i = 0; i < 16; ++i) acc[i] = zero;

  auto loadA = [&](const _Float16* mb, int stride, int mt, int kc32) -> v16h {
    int k0 = (lane >= 16) ? 8 : 0;
    const _Float16* p0 = mb + (mt * 16 + n16) * stride + kc32 + k0;
    union { v16h v; v8h h[2]; } u;
    u.h[0] = *(const v8h*)(p0);
    u.h[1] = *(const v8h*)(p0 + 16);
    return u.v;
  };

  __syncthreads();

  for (int h = 0; h < HEADS; ++h) {
    // ---- TDM: stage this head's 128KB of packed weight fragments into LDS.
    // D#: 1D tile, data_size=8B, tile_dim0=tensor_dim0=16384, type=2, count=1.
    if (wave == 0) {
      unsigned long long ga =
          (unsigned long long)(const void*)pack + (unsigned long long)h * HBYTES;
      u32x4 g0 = { 1u,                                  // count=1 (valid D#)
                   (unsigned)OFF_WB,                    // lds_addr
                   (unsigned)(ga & 0xFFFFFFFFu),        // global_addr[31:0]
                   (unsigned)((ga >> 32) & 0x01FFFFFFu) | 0x80000000u }; // +type=2
      i32x8 g1 = { (int)(3u << 16),     // data_size=8B; no multicast/pad/iterate
                   (int)0x40000000,     // tensor_dim0[15:0]=16384 in bits[63:48]
                   (int)0x00010000,     // tensor_dim1=1 in bits[95:80]
                   (int)0x40000000,     // tile_dim0=16384 in bits[127:112]
                   0,                   // tile_dim1=0 (unused), tile_dim2=0
                   16384, 0, 0 };       // tensor_dim0_stride=16384
      v4i  gz4 = {};
      i32x8 gz8 = {};
      __builtin_amdgcn_tensor_load_to_lds(g0, g1, gz4, gz4, gz8, 0);
    }

    // overlap TDM with A-fragment preload (xl is stable)
    v16h a8[8];
    #pragma unroll
    for (int kc = 0; kc < 8; ++kc) a8[kc] = loadA(xl, XS, wave, kc * 32);

    if (wave == 0) __builtin_amdgcn_s_wait_tensorcnt(0);
    __syncthreads();                       // weights visible to all waves

    // ---- QKV: wave owns m-tile `wave`; A-frags cached in regs, reused 12x ----
    #pragma unroll
    for (int mat = 0; mat < 3; ++mat) {
      const _Float16* chunk = ws + mat * HCHUNK;
      #pragma unroll
      for (int nt4 = 0; nt4 < 4; ++nt4) {
        v8f c = zero;
        #pragma unroll
        for (int kc = 0; kc < 8; ++kc) {
          v16h b = *(const v16h*)(chunk + (((nt4 * 8 + kc) << 9) + (lane << 4)));
          c = wmma32(a8[kc], b, c);
        }
        if (mat == 0) {                     // Q -> row-major (A-frag source)
          #pragma unroll
          for (int rr = 0; rr < 8; ++rr)
            ql[(wave * 16 + mofs + rr) * QS + nt4 * 16 + n16] = (_Float16)c[rr];
        } else if (mat == 1) {              // K -> B-fragment layout (k=d, n=token)
          int d0 = nt4 * 16 + n16;
          int kc2 = d0 >> 5, kf = d0 & 31;
          int lb0 = (kf & 8) ? 16 : 0;
          int e   = ((kf & 16) ? 8 : 0) + (kf & 7);
          #pragma unroll
          for (int rr = 0; rr < 8; ++rr)
            kb[((wave * 2 + kc2) * 32 + lb0 + mofs + rr) * 16 + e] = (_Float16)c[rr];
        } else {                            // V -> B-fragment layout (k=token, n=d)
          v8h hv;                           // lane slot fixed, e=rr: one b128 store
          #pragma unroll
          for (int rr = 0; rr < 8; ++rr) hv[rr] = (_Float16)c[rr];
          int lb = (mofs ? 16 : 0) + n16;
          *(v8h*)(vb + ((wave * 4 + nt4) * 32 + lb) * 16) = hv;
        }
      }
    }
    __syncthreads();

    // ---- S = Q K^T per 16-row group (2 seqs); wave w = group w ----
    {
      v8f c = zero;
      #pragma unroll
      for (int kc = 0; kc < 2; ++kc) {
        v16h a = loadA(ql, QS, wave, kc * 32);
        v16h b = *(const v16h*)(kb + ((wave * 2 + kc) * 32 + lane) * 16);
        c = wmma32(a, b, c);
      }
      #pragma unroll
      for (int rr = 0; rr < 8; ++rr)
        sS[wave * 256 + (mofs + rr) * 16 + n16] = c[rr];
    }
    __syncthreads();

    // ---- tiny row softmax (128 threads, 7-wide), write P as A-fragments ----
    if (tid < MR) {
      int grp = tid >> 4, i = tid & 15;
      int jb = i & 8;                       // own-sequence column block
      float d[PATCH];
      #pragma unroll
      for (int jj = 0; jj < PATCH; ++jj)
        d[jj] = sS[grp * 256 + i * 16 + jb + jj] * 0.125f;
      float mx = d[0];
      #pragma unroll
      for (int jj = 1; jj < PATCH; ++jj) mx = fmaxf(mx, d[jj]);
      float w[PATCH], sum = 0.f;
      #pragma unroll
      for (int jj = 0; jj < PATCH; ++jj) { w[jj] = __expf(d[jj] - mx); sum += w[jj]; }
      float inv = 1.f / sum;
      int L = (i < 8) ? i : (i + 16);
      _Float16* pdst = pa + (grp * 32 + L) * 16;
      #pragma unroll
      for (int jj = 0; jj < PATCH; ++jj) pdst[jj] = (_Float16)(w[jj] * inv);
    }
    __syncthreads();

    // ---- O = P V per group (K=32, upper half statically zero) ----
    {
      v16h a = *(const v16h*)(pa + (wave * 32 + lane) * 16);
      v8f oc[4];
      #pragma unroll
      for (int nt = 0; nt < 4; ++nt) {      // 4 independent WMMAs first,
        v16h b = *(const v16h*)(vb + ((wave * 4 + nt) * 32 + lane) * 16);
        oc[nt] = wmma32(a, b, zero);
      }
      #pragma unroll
      for (int nt = 0; nt < 4; ++nt) {      // ... then D readback (hides hazard)
        #pragma unroll
        for (int rr = 0; rr < 8; ++rr)
          ol[(wave * 16 + mofs + rr) * QS + nt * 16 + n16] = (_Float16)oc[nt][rr];
      }
    }
    __syncthreads();

    // ---- y += O_h @ Wo_h^T; 2 A-frags reused across 16 n-tiles ----
    {
      const _Float16* chunk = ws + 3 * HCHUNK;
      v16h aP0 = loadA(ol, QS, wave, 0);
      v16h aP1 = loadA(ol, QS, wave, 32);
      #pragma unroll
      for (int nt = 0; nt < 16; ++nt) {
        v16h b0 = *(const v16h*)(chunk + (((nt * 2 + 0) << 9) + (lane << 4)));
        acc[nt] = wmma32(aP0, b0, acc[nt]);
        v16h b1 = *(const v16h*)(chunk + (((nt * 2 + 1) << 9) + (lane << 4)));
        acc[nt] = wmma32(aP1, b1, acc[nt]);
      }
    }
    __syncthreads();
  }

  // ---- transpose y through LDS (aliases tile memory), coalesced store ----
  #pragma unroll
  for (int nt = 0; nt < 16; ++nt) {
    #pragma unroll
    for (int rr = 0; rr < 8; ++rr)
      ystage[(wave * 16 + mofs + rr) * YS + nt * 16 + n16] = acc[nt][rr];
  }
  __syncthreads();

  for (int pass = 0; pass < 112; ++pass) {
    int pair = pass * 16 + (tid >> 4);      // pair = o*7 + p  (pad rows skipped)
    int o = pair / 7, p = pair % 7;
    float y = ystage[(sl * PPAD + p) * YS + o] + bo[o];
    out[gbase + (long)o * (TTOT * VDIM) + p * VDIM] = y;
  }
}

extern "C" void kernel_launch(void* const* d_in, const int* in_sizes, int n_in,
                              void* d_out, int out_size, void* d_ws, size_t ws_size,
                              hipStream_t stream) {
  const float* fmap = (const float*)d_in[0];
  const float* Wq   = (const float*)d_in[1];
  const float* Wkv  = (const float*)d_in[2];
  const float* Wo   = (const float*)d_in[3];
  const float* bo   = (const float*)d_in[4];
  float* outp = (float*)d_out;
  _Float16* pk = (_Float16*)d_ws;           // 1 MB packed-weight scratch

  pack_weights_kernel<<<2048, 256, 0, stream>>>(Wq, Wkv, Wo, pk);
  local_attn_kernel<<<NGRP, 256, SMEM_BYTES, stream>>>(fmap, pk, bo, outp);
}
